// NonLocalLayer_47974784696817
// MI455X (gfx1250) — compile-verified, matrix operand order checked by reading, not run on hardware
//
#include <hip/hip_runtime.h>
#include <hip/hip_bf16.h>

// ---------------------------------------------------------------------------
// NonLocal (embedded-gaussian) block, MI455X / gfx1250.
// bf16 WMMA (v_wmma_f32_16x16x32_bf16) for every GEMM, flash-style fused
// attention (no [B,N,N] tensor in HBM), BN stats fused into the output GEMM.
// B=8, C=512, N=2048, P=256.
// ---------------------------------------------------------------------------

typedef __bf16 v16bf __attribute__((ext_vector_type(16)));
typedef float  v8f   __attribute__((ext_vector_type(8)));
typedef unsigned short u16x8 __attribute__((ext_vector_type(8)));

static constexpr int Bq = 8;     // batch
static constexpr int Cc = 512;   // channels
static constexpr int Nn = 2048;  // sequence length
static constexpr int Pp = 256;   // inner channels

__device__ __forceinline__ unsigned short f2bf(float f) {
    union { float f; unsigned u; } v; v.f = f;
    unsigned r = v.u + 0x7FFFu + ((v.u >> 16) & 1u);   // round-to-nearest-even
    return (unsigned short)(r >> 16);
}

__device__ __forceinline__ v16bf ldfrag(const unsigned short* p) {
    return *reinterpret_cast<const v16bf*>(p);         // 32B contiguous frag
}

// ---------------------------------------------------------------------------
// Kernel 1: fp32 -> bf16 conversion of x and all weights; zero BN stat bufs.
// ---------------------------------------------------------------------------
__global__ void k_convert(const float* __restrict__ x,
                          const float* __restrict__ Wg,
                          const float* __restrict__ Wt,
                          const float* __restrict__ Wp,
                          const float* __restrict__ Wz,
                          unsigned short* __restrict__ x_bf,
                          unsigned short* __restrict__ W_all,   // [3P][C]
                          unsigned short* __restrict__ Wz_bf,   // [C][P]
                          float* __restrict__ sums, float* __restrict__ sumsq) {
    const long long XN = (long long)Bq * Cc * Nn;
    const long long WN = (long long)Pp * Cc;
    const long long ZW = (long long)Cc * Pp;
    const long long total = XN + 3 * WN + ZW + Cc;
    for (long long i = blockIdx.x * (long long)blockDim.x + threadIdx.x;
         i < total; i += (long long)gridDim.x * blockDim.x) {
        if (i < XN)               x_bf[i] = f2bf(x[i]);
        else if (i < XN + WN)     W_all[i - XN] = f2bf(Wg[i - XN]);
        else if (i < XN + 2*WN)   W_all[i - XN] = f2bf(Wt[i - XN - WN]);
        else if (i < XN + 3*WN)   W_all[i - XN] = f2bf(Wp[i - XN - 2*WN]);
        else if (i < XN + 3*WN + ZW) Wz_bf[i - XN - 3*WN] = f2bf(Wz[i - XN - 3*WN]);
        else { int c = (int)(i - XN - 3*WN - ZW); sums[c] = 0.f; sumsq[c] = 0.f; }
    }
}

// ---------------------------------------------------------------------------
// Kernel 2: fused projection GEMM. [3P x C] * [C x N] per batch.
// Block = 128x128 tile, 8 waves (4x2), each wave 32x64 = 8 wmma accumulators.
// Output routing is decided by blockIdx.y (scalar-uniform -> SALU branch):
//   region 0: g     -> [B][P][N]  (scalar b16 stores, transposed layout)
//   region 1: theta -> [B][N][P]  (one b128 store per 16x16 tile half)
//   region 2: phi   -> [B][N][P]  (one b128 store per 16x16 tile half)
// ---------------------------------------------------------------------------
__global__ void k_proj(const unsigned short* __restrict__ x_bf,
                       const unsigned short* __restrict__ W_all,
                       unsigned short* __restrict__ gbuf,
                       unsigned short* __restrict__ tbuf,
                       unsigned short* __restrict__ pbuf) {
    __shared__ __align__(32) unsigned short ldsA[128 * 32];  // [j][k]
    __shared__ __align__(32) unsigned short ldsB[128 * 32];  // [n][k] (transposed x)
    const int b  = blockIdx.z;
    const int n0 = blockIdx.x * 128;
    const int j0 = blockIdx.y * 128;
    const int tid = threadIdx.x, lane = tid & 31, w = tid >> 5;
    const int wr = w & 3, wc = w >> 2;
    const int hi = lane >> 4, lo = lane & 15;

    v8f acc[2][4];
#pragma unroll
    for (int a = 0; a < 2; ++a)
#pragma unroll
        for (int c = 0; c < 4; ++c) acc[a][c] = {};

    for (int c0 = 0; c0 < Cc; c0 += 32) {
        // stage A (weights, row-major) and B (x, transposed to [n][k])
        const int r = tid & 127, kg = tid >> 7;
#pragma unroll
        for (int i = 0; i < 16; ++i) {
            const int k = kg * 16 + i;
            ldsA[r * 32 + k] = W_all[(long long)(j0 + r) * Cc + c0 + k];
            ldsB[r * 32 + k] = x_bf[(long long)b * Cc * Nn + (long long)(c0 + k) * Nn + n0 + r];
        }
        __syncthreads();

        v16bf af[2], bfv[4];
#pragma unroll
        for (int sr = 0; sr < 2; ++sr)
            af[sr] = ldfrag(&ldsA[(wr * 32 + sr * 16 + lo) * 32 + hi * 16]);
#pragma unroll
        for (int sc = 0; sc < 4; ++sc)
            bfv[sc] = ldfrag(&ldsB[(wc * 64 + sc * 16 + lo) * 32 + hi * 16]);
#pragma unroll
        for (int sr = 0; sr < 2; ++sr)
#pragma unroll
            for (int sc = 0; sc < 4; ++sc)
                acc[sr][sc] = __builtin_amdgcn_wmma_f32_16x16x32_bf16(
                    false, af[sr], false, bfv[sc], (short)0, acc[sr][sc], false, false);
        __syncthreads();
    }

    const int region = j0 >> 8;   // 0:g  1:theta  2:phi   (block-uniform, SGPR)
    if (region == 0) {
        // g[b][j][n]: j varies with accumulator row -> scalar b16 stores
#pragma unroll
        for (int sr = 0; sr < 2; ++sr)
#pragma unroll
            for (int sc = 0; sc < 4; ++sc)
#pragma unroll
                for (int v = 0; v < 8; ++v) {
                    const int j = j0 + wr * 32 + sr * 16 + ((lane < 16) ? v : v + 8);
                    const int n = n0 + wc * 64 + sc * 16 + lo;
                    gbuf[(long long)b * Pp * Nn + (long long)j * Nn + n] = f2bf(acc[sr][sc][v]);
                }
    } else {
        // theta/phi in [B][N][P]: lane's 8 values are contiguous in p -> b128
        unsigned short* dst = (region == 1 ? tbuf : pbuf) + (long long)b * Nn * Pp;
        const int pbase = (j0 - region * Pp) + wr * 32;
#pragma unroll
        for (int sr = 0; sr < 2; ++sr)
#pragma unroll
            for (int sc = 0; sc < 4; ++sc) {
                const int n = n0 + wc * 64 + sc * 16 + lo;
                const int p0 = pbase + sr * 16 + ((lane < 16) ? 0 : 8);
                u16x8 pk;
#pragma unroll
                for (int v = 0; v < 8; ++v) pk[v] = f2bf(acc[sr][sc][v]);
                *reinterpret_cast<u16x8*>(&dst[(long long)n * Pp + p0]) = pk;
            }
    }
}

// ---------------------------------------------------------------------------
// Kernel 3: flash attention. Block = 4 waves = 64 query rows (more blocks ->
// better WGP spread), wave = 16 rows. theta fragments pinned in registers;
// online softmax with 16-lane shuffle reductions; probability tile transposed
// D->A layout through a wave-private LDS buffer (explicit s_wait_dscnt);
// y accumulated over all 16 p-tiles with WMMA. 2048 wmmas per wave.
// ---------------------------------------------------------------------------
__global__ void k_attn(const unsigned short* __restrict__ gbuf,
                       const unsigned short* __restrict__ tbuf,
                       const unsigned short* __restrict__ pbuf,
                       unsigned short* __restrict__ ybuf) {
    __shared__ __align__(32) unsigned short probs[4][16 * 32];
    const int b  = blockIdx.y;
    const int n0 = blockIdx.x * 64;
    const int tid = threadIdx.x, lane = tid & 31, w = tid >> 5;
    const int hi = lane >> 4, lo = lane & 15;
    const int rowbase = n0 + w * 16;
    const unsigned short* tb = tbuf + (long long)b * Nn * Pp;
    const unsigned short* pb = pbuf + (long long)b * Nn * Pp;
    const unsigned short* gb = gbuf + (long long)b * Pp * Nn;

    v16bf tf[8];
#pragma unroll
    for (int kk = 0; kk < 8; ++kk)
        tf[kk] = ldfrag(&tb[(long long)(rowbase + lo) * Pp + kk * 32 + hi * 16]);

    v8f yacc[16];
#pragma unroll
    for (int pt = 0; pt < 16; ++pt) yacc[pt] = {};
    float rmax[8], rsum[8];
#pragma unroll
    for (int v = 0; v < 8; ++v) { rmax[v] = -3.0e38f; rsum[v] = 0.0f; }

    unsigned short* pw = probs[w];

    for (int m0 = 0; m0 < Nn; m0 += 32) {
        v8f s0 = {}, s1 = {};
#pragma unroll
        for (int kk = 0; kk < 8; ++kk) {
            v16bf b0 = ldfrag(&pb[(long long)(m0 +      lo) * Pp + kk * 32 + hi * 16]);
            v16bf b1 = ldfrag(&pb[(long long)(m0 + 16 + lo) * Pp + kk * 32 + hi * 16]);
            s0 = __builtin_amdgcn_wmma_f32_16x16x32_bf16(false, tf[kk], false, b0, (short)0, s0, false, false);
            s1 = __builtin_amdgcn_wmma_f32_16x16x32_bf16(false, tf[kk], false, b1, (short)0, s1, false, false);
        }
        if (m0 + 32 < Nn) {  // pull next phi tile toward the WGP
            __builtin_prefetch(&pb[(long long)(m0 + 32 + lo) * Pp], 0, 3);
            __builtin_prefetch(&gb[(long long)lo * Nn + m0 + 32], 0, 3);
        }

        // online softmax update (row stats replicated across each 16-lane half)
        float corr[8];
#pragma unroll
        for (int v = 0; v < 8; ++v) {
            float mt = fmaxf(s0[v], s1[v]);
#pragma unroll
            for (int off = 1; off < 16; off <<= 1) mt = fmaxf(mt, __shfl_xor(mt, off, 16));
            const float mnew = fmaxf(rmax[v], mt);
            const float p0 = __expf(s0[v] - mnew), p1 = __expf(s1[v] - mnew);
            float ls = p0 + p1;
#pragma unroll
            for (int off = 1; off < 16; off <<= 1) ls += __shfl_xor(ls, off, 16);
            corr[v]  = __expf(rmax[v] - mnew);
            rsum[v]  = rsum[v] * corr[v] + ls;
            rmax[v]  = mnew;
            s0[v] = p0; s1[v] = p1;
        }
#pragma unroll
        for (int pt = 0; pt < 16; ++pt)
#pragma unroll
            for (int v = 0; v < 8; ++v) yacc[pt][v] *= corr[v];

        // D-layout probs -> A-layout fragment via wave-private LDS
#pragma unroll
        for (int v = 0; v < 8; ++v) {
            const int row = (lane < 16) ? v : v + 8;
            pw[row * 32 +      lo] = f2bf(s0[v]);
            pw[row * 32 + 16 + lo] = f2bf(s1[v]);
        }
        asm volatile("s_wait_dscnt 0x0" ::: "memory");
        v16bf pa = ldfrag(&pw[lo * 32 + hi * 16]);

#pragma unroll
        for (int pt = 0; pt < 16; ++pt) {
            v16bf gf = ldfrag(&gb[(long long)(pt * 16 + lo) * Nn + m0 + hi * 16]);
            yacc[pt] = __builtin_amdgcn_wmma_f32_16x16x32_bf16(false, pa, false, gf, (short)0, yacc[pt], false, false);
        }
    }

    // normalize by row sums, store y in [B][N][P] (z-GEMM B-fragment layout)
#pragma unroll
    for (int pt = 0; pt < 16; ++pt)
#pragma unroll
        for (int v = 0; v < 8; ++v) {
            const int n = rowbase + ((lane < 16) ? v : v + 8);
            const int p = pt * 16 + lo;
            ybuf[(long long)b * Nn * Pp + (long long)n * Pp + p] = f2bf(yacc[pt][v] / rsum[v]);
        }
}

// ---------------------------------------------------------------------------
// Kernel 4: z = Wz * y (per batch) with fused BN statistics. Both operands
// already fragment-contiguous -> no LDS staging needed. Stats reduced across
// the 16-lane row group then one atomicAdd per row per group.
// ---------------------------------------------------------------------------
__global__ void k_zgemm(const unsigned short* __restrict__ Wz_bf,
                        const unsigned short* __restrict__ ybuf,
                        float* __restrict__ zout,
                        float* __restrict__ sums, float* __restrict__ sumsq) {
    const int b  = blockIdx.z;
    const int n0 = blockIdx.x * 128;
    const int c0 = blockIdx.y * 128;
    const int tid = threadIdx.x, lane = tid & 31, w = tid >> 5;
    const int wr = w & 3, wc = w >> 2;
    const int hi = lane >> 4, lo = lane & 15;
    const unsigned short* yb = ybuf + (long long)b * Nn * Pp;

    v8f acc[2][4];
#pragma unroll
    for (int a = 0; a < 2; ++a)
#pragma unroll
        for (int c = 0; c < 4; ++c) acc[a][c] = {};

#pragma unroll
    for (int kk = 0; kk < 8; ++kk) {
        v16bf af[2], bfv[4];
#pragma unroll
        for (int sr = 0; sr < 2; ++sr)
            af[sr] = ldfrag(&Wz_bf[(long long)(c0 + wr * 32 + sr * 16 + lo) * Pp + kk * 32 + hi * 16]);
#pragma unroll
        for (int sc = 0; sc < 4; ++sc)
            bfv[sc] = ldfrag(&yb[(long long)(n0 + wc * 64 + sc * 16 + lo) * Pp + kk * 32 + hi * 16]);
#pragma unroll
        for (int sr = 0; sr < 2; ++sr)
#pragma unroll
            for (int sc = 0; sc < 4; ++sc)
                acc[sr][sc] = __builtin_amdgcn_wmma_f32_16x16x32_bf16(
                    false, af[sr], false, bfv[sc], (short)0, acc[sr][sc], false, false);
    }

#pragma unroll
    for (int sr = 0; sr < 2; ++sr)
#pragma unroll
        for (int v = 0; v < 8; ++v) {
            const int c = c0 + wr * 32 + sr * 16 + ((lane < 16) ? v : v + 8);
            float rs = 0.f, rq = 0.f;
#pragma unroll
            for (int sc = 0; sc < 4; ++sc) {
                const int n = n0 + wc * 64 + sc * 16 + lo;
                const float z = acc[sr][sc][v];
                zout[(long long)b * Cc * Nn + (long long)c * Nn + n] = z;
                rs += z; rq += z * z;
            }
#pragma unroll
            for (int off = 1; off < 16; off <<= 1) {
                rs += __shfl_xor(rs, off, 16);
                rq += __shfl_xor(rq, off, 16);
            }
            if (lo == 0) { atomicAdd(&sums[c], rs); atomicAdd(&sumsq[c], rq); }
        }
}

// ---------------------------------------------------------------------------
// Kernel 5: BatchNorm finalize + residual (in-place on d_out).
// ---------------------------------------------------------------------------
__global__ void k_bn(const float* __restrict__ x,
                     const float* __restrict__ gamma,
                     const float* __restrict__ beta,
                     const float* __restrict__ sums,
                     const float* __restrict__ sumsq,
                     float* __restrict__ out) {
    const float invMN = 1.0f / (float)(Bq * Nn);
    const long long total = (long long)Bq * Cc * Nn;
    for (long long i = blockIdx.x * (long long)blockDim.x + threadIdx.x;
         i < total; i += (long long)gridDim.x * blockDim.x) {
        const int c = (int)((i / Nn) % Cc);
        const float mean = sums[c] * invMN;
        const float var  = sumsq[c] * invMN - mean * mean;
        const float z = out[i];
        out[i] = (z - mean) * rsqrtf(var + 1e-5f) * gamma[c] + beta[c] + x[i];
    }
}

// ---------------------------------------------------------------------------
extern "C" void kernel_launch(void* const* d_in, const int* in_sizes, int n_in,
                              void* d_out, int out_size, void* d_ws, size_t ws_size,
                              hipStream_t stream) {
    const float* x     = (const float*)d_in[0];
    const float* Wg    = (const float*)d_in[1];
    const float* Wt    = (const float*)d_in[2];
    const float* Wp    = (const float*)d_in[3];
    const float* Wz    = (const float*)d_in[4];
    const float* gamma = (const float*)d_in[5];
    const float* beta  = (const float*)d_in[6];
    float* out = (float*)d_out;

    char* ws = (char*)d_ws;
    size_t off = 0;
    auto carve = [&](size_t bytes) -> char* {
        off = (off + 255) & ~(size_t)255;
        char* p = ws + off;
        off += bytes;
        return p;
    };
    unsigned short* x_bf  = (unsigned short*)carve((size_t)Bq * Cc * Nn * 2);
    unsigned short* W_all = (unsigned short*)carve((size_t)3 * Pp * Cc * 2);
    unsigned short* Wz_bf = (unsigned short*)carve((size_t)Cc * Pp * 2);
    unsigned short* gbuf  = (unsigned short*)carve((size_t)Bq * Pp * Nn * 2);
    unsigned short* tbuf  = (unsigned short*)carve((size_t)Bq * Nn * Pp * 2);
    unsigned short* pbuf  = (unsigned short*)carve((size_t)Bq * Nn * Pp * 2);
    unsigned short* ybuf  = (unsigned short*)carve((size_t)Bq * Nn * Pp * 2);
    float* sums  = (float*)carve((size_t)Cc * 4);
    float* sumsq = (float*)carve((size_t)Cc * 4);

    k_convert<<<dim3(4096), dim3(256), 0, stream>>>(x, Wg, Wt, Wp, Wz,
                                                    x_bf, W_all, Wz_bf, sums, sumsq);
    k_proj<<<dim3(Nn / 128, (3 * Pp) / 128, Bq), dim3(256), 0, stream>>>(
        x_bf, W_all, gbuf, tbuf, pbuf);
    k_attn<<<dim3(Nn / 64, Bq), dim3(128), 0, stream>>>(gbuf, tbuf, pbuf, ybuf);
    k_zgemm<<<dim3(Nn / 128, Cc / 128, Bq), dim3(256), 0, stream>>>(
        Wz_bf, ybuf, out, sums, sumsq);
    k_bn<<<dim3(4096), dim3(256), 0, stream>>>(x, gamma, beta, sums, sumsq, out);
}
